// HaplotypeEmbedding_87840671138034
// MI455X (gfx1250) — compile-verified, matrix operand order checked by reading, not run on hardware
//
#include <hip/hip_runtime.h>

// ---------------------------------------------------------------------------
// Fused haplotype-embedding + 2-layer MLP for MI455X (gfx1250, wave32, WMMA).
//   prep A: W1/W2 -> f16, swizzled into WMMA B-fragment layout in d_ws
//   prep B: tables -> f16 copy (padding row zeroed) in d_ws  [gather = pure DMA]
//   stage 1: async gather (global_load_async_to_lds_b128, double-buffered LDS)
//            overlapped with h = x@W1+b1 WMMAs
//   epilogue: exact-erf GELU, h -> LDS as f16
//   stage 2: out = h@W2 + b2, f32 stores
// ---------------------------------------------------------------------------

typedef __attribute__((ext_vector_type(16))) _Float16 v16h;
typedef __attribute__((ext_vector_type(8)))  float    v8f;

#define LOCI 8
#define DIM  256
#define VOC  512
#define HID  512
#define LD   2048          // LOCI*DIM (GEMM1 K)
#define MT   64            // rows per workgroup
#define XS_STRIDE 264      // f16 stride for x chunk (256 + 8 pad, 16B aligned)
#define HS_STRIDE 520      // f16 stride for h      (512 + 8 pad, 16B aligned)

union FragH { uint4 q[2]; v16h v; };
static_assert(sizeof(v16h) == 32, "v16h must be 8 VGPRs");

__device__ __forceinline__ unsigned int pack_f16(float a, float b) {
  union { _Float16 h[2]; unsigned int u; } p;
  p.h[0] = (_Float16)a;
  p.h[1] = (_Float16)b;
  return p.u;
}

__device__ __forceinline__ unsigned short f16_bits(float a) {
  union { _Float16 h; unsigned short s; } p;
  p.h = (_Float16)a;
  return p.s;
}

// CDNA5 async global->LDS copy, 16B per lane, tracked with ASYNCcnt.
__device__ __forceinline__ void async_copy_b128(unsigned lds_addr,
                                                const void* gaddr) {
  asm volatile("global_load_async_to_lds_b128 %0, %1, off"
               :: "v"(lds_addr), "v"(gaddr)
               : "memory");
}

__device__ __forceinline__ void wait_asynccnt0() {
#if __has_builtin(__builtin_amdgcn_s_wait_asynccnt)
  __builtin_amdgcn_s_wait_asynccnt(0);
#else
  asm volatile("s_wait_asynccnt 0x0" ::: "memory");
#endif
}

// ---------------------------------------------------------------------------
// Prep: swizzle a row-major [K, ncols] f32 weight matrix into per-block WMMA
// B-fragment layout (f16). Block = 32(K) x 16(N); within a block dwords are
// [chunk(2)][lane(32)][4 dwords] so the consumer does two coalesced b128 loads.
// B-operand VGPR mapping (mirror of the 16-bit A layout, ISA 7.12.2):
//   n = lane%16 ;  k0 = (v/4)*16 + (lane/16)*8 + (v%4)*2 ; pair (k0, k0+1)
// ---------------------------------------------------------------------------
__global__ __launch_bounds__(256) void swizzle_w_kernel(
    const float* __restrict__ W, unsigned int* __restrict__ dst,
    int kblocks, int ncols) {
  const int gid = blockIdx.x * 256 + threadIdx.x;
  const int nblkCount = ncols >> 4;
  const int total = kblocks * nblkCount * 256;
  if (gid >= total) return;
  const int blk    = gid >> 8;
  const int within = gid & 255;
  const int c    = within >> 7;
  const int lane = (within >> 2) & 31;
  const int vv   = within & 3;
  const int v    = c * 4 + vv;
  const int kblk = blk / nblkCount;
  const int nblk = blk % nblkCount;
  const int n  = nblk * 16 + (lane & 15);
  const int k0 = (v >> 2) * 16 + ((lane >> 4)) * 8 + (v & 3) * 2;
  const int k  = kblk * 32 + k0;
  dst[gid] = pack_f16(W[(size_t)k * ncols + n], W[(size_t)(k + 1) * ncols + n]);
}

// ---------------------------------------------------------------------------
// Prep: tables f32 -> f16 (padding row tok==0 forced to zero). After this the
// per-locus gather is pure data movement (512B rows), done with async DMA.
// ---------------------------------------------------------------------------
__global__ __launch_bounds__(256) void tables_f16_kernel(
    const float* __restrict__ T, unsigned int* __restrict__ dst,
    int totalPairs) {
  const int gid = blockIdx.x * 256 + threadIdx.x;
  if (gid >= totalPairs) return;
  float a = T[(size_t)gid * 2 + 0];
  float b = T[(size_t)gid * 2 + 1];
  const int row = (gid >> 7) & (VOC - 1);  // 128 dwords per 256-f32 row
  if (row == 0) { a = 0.0f; b = 0.0f; }
  dst[gid] = pack_f16(a, b);
}

// ---------------------------------------------------------------------------
// Main fused kernel. 256 threads = 8 wave32. M-tile = 64 rows.
// Stage-1 wave mapping: rowblk = wave&3 (16 rows), colgrp = wave>>2 (256 cols).
// Stage-2 wave mapping: rowblk, colsl = wave>>2 (128 cols of DIM).
// ---------------------------------------------------------------------------
__global__ __launch_bounds__(256) void hap_mlp_kernel(
    const int* __restrict__ hap, const unsigned short* __restrict__ tf16,
    const float* __restrict__ b1, const float* __restrict__ b2,
    const unsigned int* __restrict__ w1s, const unsigned int* __restrict__ w2s,
    float* __restrict__ out) {
  __shared__ __align__(16) unsigned short xs[2][MT * XS_STRIDE];
  __shared__ __align__(16) unsigned short hs[MT * HS_STRIDE];

  const int tid  = threadIdx.x;
  const int lane = tid & 31;
  const int wave = tid >> 5;
  const int row0 = blockIdx.x * MT;

  const int rowblk = wave & 3;
  const int colgrp = wave >> 2;
  const int nbase  = colgrp * 256;
  const int ln = lane & 15;
  const int kh = (lane >> 4) * 8;

  const uint4* w1q = (const uint4*)w1s;
  const uint4* w2q = (const uint4*)w2s;

  // ------ stage 1 accumulators, seeded with b1 ------
  v8f acc[16];
  #pragma unroll
  for (int t = 0; t < 16; ++t) {
    const float bv = b1[nbase + t * 16 + ln];
    #pragma unroll
    for (int r = 0; r < 8; ++r) acc[t][r] = bv;
  }

  // gather mapping: 4 threads per row, 128B chunk each (8 async b128 per thread)
  const int gr = tid >> 2;
  const int gq = tid & 3;
  auto issue_gather = [&](int l, int buf) {
    int tok = hap[(row0 + gr) * LOCI + l];
    tok = tok < 0 ? 0 : (tok > VOC - 1 ? VOC - 1 : tok);
    const unsigned short* src =
        tf16 + ((size_t)(l * VOC + tok)) * DIM + gq * 64;
    const unsigned lbase =
        (unsigned)(unsigned long long)(&xs[buf][gr * XS_STRIDE + gq * 64]);
    #pragma unroll
    for (int i = 0; i < 8; ++i)
      async_copy_b128(lbase + i * 16, src + i * 8);
  };

  issue_gather(0, 0);

  #pragma unroll 1
  for (int l = 0; l < LOCI; ++l) {
    wait_asynccnt0();      // this wave's copies into xs[l&1] complete
    __syncthreads();       // all waves' copies visible; prev compute done
    if (l + 1 < LOCI) issue_gather(l + 1, (l + 1) & 1);  // overlap next gather

    // ---- GEMM1 partial: K = 256 for this locus, steps of 32
    const unsigned short* arow =
        &xs[l & 1][(rowblk * 16 + ln) * XS_STRIDE + kh];
    for (int kk = 0; kk < 256; kk += 32) {
      FragH a;
      a.q[0] = *(const uint4*)(arow + kk);
      a.q[1] = *(const uint4*)(arow + kk + 16);
      const int kblk = l * 8 + (kk >> 5);
      #pragma unroll
      for (int t = 0; t < 16; ++t) {
        const int blk = kblk * 32 + colgrp * 16 + t;
        FragH b;
        b.q[0] = w1q[blk * 64 + lane];
        b.q[1] = w1q[blk * 64 + 32 + lane];
        acc[t] = __builtin_amdgcn_wmma_f32_16x16x32_f16(
            false, a.v, false, b.v, (short)0, acc[t], false, false);
      }
    }
  }
  __syncthreads();

  // ------ epilogue 1: exact GELU (erf), f16 into LDS h buffer ------
  #pragma unroll
  for (int t = 0; t < 16; ++t) {
    #pragma unroll
    for (int r = 0; r < 8; ++r) {
      const float x = acc[t][r];
      const float g = 0.5f * x * (1.0f + erff(x * 0.70710678118654752f));
      const int mrow = rowblk * 16 + r + (lane >> 4) * 8;
      const int col  = nbase + t * 16 + ln;
      hs[mrow * HS_STRIDE + col] = f16_bits(g);
    }
  }
  __syncthreads();

  // ------ stage 2: out = h @ W2 + b2 ------
  const int colsl = wave >> 2;
  const int obase = colsl * 128;
  v8f acc2[8];
  #pragma unroll
  for (int t = 0; t < 8; ++t) {
    const float bv = b2[obase + t * 16 + ln];
    #pragma unroll
    for (int r = 0; r < 8; ++r) acc2[t][r] = bv;
  }

  const unsigned short* hrow = hs + (rowblk * 16 + ln) * HS_STRIDE + kh;
  for (int kk = 0; kk < HID; kk += 32) {
    FragH a;
    a.q[0] = *(const uint4*)(hrow + kk);
    a.q[1] = *(const uint4*)(hrow + kk + 16);
    const int kblk = kk >> 5;
    #pragma unroll
    for (int t = 0; t < 8; ++t) {
      const int blk = kblk * 16 + colsl * 8 + t;
      FragH b;
      b.q[0] = w2q[blk * 64 + lane];
      b.q[1] = w2q[blk * 64 + 32 + lane];
      acc2[t] = __builtin_amdgcn_wmma_f32_16x16x32_f16(
          false, a.v, false, b.v, (short)0, acc2[t], false, false);
    }
  }

  // ------ store: f32, coalesced 64B runs per half-wave ------
  #pragma unroll
  for (int t = 0; t < 8; ++t) {
    #pragma unroll
    for (int r = 0; r < 8; ++r) {
      const int mrow = rowblk * 16 + r + (lane >> 4) * 8;
      const int col  = obase + t * 16 + ln;
      out[(size_t)(row0 + mrow) * DIM + col] = acc2[t][r];
    }
  }
}

// ---------------------------------------------------------------------------
// Launch. Input order: haplotypes, tables, W1, b1, W2, b2.
// d_ws layout: [0,2MB) f16 W1 swizzled ; [2MB,2.25MB) f16 W2 swizzled ;
//              [2.25MB,4.25MB) f16 tables.
// ---------------------------------------------------------------------------
extern "C" void kernel_launch(void* const* d_in, const int* in_sizes, int n_in,
                              void* d_out, int out_size, void* d_ws,
                              size_t ws_size, hipStream_t stream) {
  const int*   hap    = (const int*)d_in[0];
  const float* tables = (const float*)d_in[1];
  const float* W1     = (const float*)d_in[2];
  const float* b1v    = (const float*)d_in[3];
  const float* W2     = (const float*)d_in[4];
  const float* b2v    = (const float*)d_in[5];
  float* outp = (float*)d_out;

  unsigned int* w1s  = (unsigned int*)d_ws;            // LD*HID/2   dwords
  unsigned int* w2s  = w1s + (LD * HID / 2);           // HID*DIM/2  dwords
  unsigned int* tf   = w2s + (HID * DIM / 2);          // LOCI*VOC*DIM/2 dwords

  const int w1_dwords = LD * HID / 2;        // 524288
  const int w2_dwords = HID * DIM / 2;       // 65536
  const int tb_dwords = LOCI * VOC * DIM / 2;// 524288
  swizzle_w_kernel<<<(w1_dwords + 255) / 256, 256, 0, stream>>>(
      W1, w1s, LD / 32, HID);
  swizzle_w_kernel<<<(w2_dwords + 255) / 256, 256, 0, stream>>>(
      W2, w2s, HID / 32, DIM);
  tables_f16_kernel<<<(tb_dwords + 255) / 256, 256, 0, stream>>>(
      tables, tf, tb_dwords);

  const int nrows = in_sizes[0] / LOCI;  // B*K = 131072
  hap_mlp_kernel<<<nrows / MT, 256, 0, stream>>>(
      hap, (const unsigned short*)tf, b1v, b2v, w1s, w2s, outp);
}
